// GRU_21895743275079
// MI455X (gfx1250) — compile-verified, matrix operand order checked by reading, not run on hardware
//
#include <hip/hip_runtime.h>
#include <hip/hip_bf16.h>
#include <math.h>

// ---- sizes ----
#define T_   512
#define B_   256
#define IND  64
#define HID  128
#define OUTD 64
#define CELL 256
#define G3   768           // 3*CELL
#define ROWS (T_ * B_)     // 131072

typedef __attribute__((ext_vector_type(16))) __bf16 v16bf;
typedef __attribute__((ext_vector_type(8)))  __bf16 v8bf;
typedef __attribute__((ext_vector_type(8)))  float  v8f;
typedef __attribute__((ext_vector_type(4)))  float  v4f;
typedef __attribute__((ext_vector_type(4)))  unsigned int u32x4;
typedef __attribute__((ext_vector_type(8)))  int i32x8;
typedef __attribute__((ext_vector_type(4)))  int i32x4;

// ---- TDM availability (device pass only; arity differs between toolchains) ----
#if defined(__AMDGCN__) && defined(__has_builtin)
#  if __has_builtin(__builtin_amdgcn_tensor_load_to_lds) && __has_builtin(__builtin_amdgcn_s_wait_tensorcnt)
#    define HAVE_TDM 1
#  endif
#endif
#ifndef HAVE_TDM
#  define HAVE_TDM 0
#endif

// ---------------- WMMA helpers ----------------
__device__ __forceinline__ v8f wmma_bf16(v16bf a, v16bf b, v8f c) {
    // (neg_a, A, neg_b, B, c_mod, C, reuse_a, reuse_b)
    return __builtin_amdgcn_wmma_f32_16x16x32_bf16(false, a, false, b, (short)0, c, false, false);
}

// A/B fragment from a row-major bf16 matrix, 16 rows starting at row_base,
// K-block of 32 starting at k0.  Layout (ISA 7.12.2, 16-bit A 16x32):
//   lane l: m = l&15, hi = l>>4 ; elems [0..7] = K k0+8*hi.. ; [8..15] = K k0+16+8*hi..
__device__ __forceinline__ v16bf frag_bf16(const __bf16* __restrict__ p, int ld,
                                           int row_base, int k0, int lane) {
    int m = lane & 15, hi = lane >> 4;
    const __bf16* r = p + (size_t)(row_base + m) * ld + k0 + hi * 8;
    v8bf lo = *(const v8bf*)(r);
    v8bf hi8 = *(const v8bf*)(r + 16);
    return __builtin_shufflevector(lo, hi8, 0,1,2,3,4,5,6,7,8,9,10,11,12,13,14,15);
}

// Same but source is f32 row-major (convert to bf16 on the fly).
__device__ __forceinline__ v16bf frag_f32(const float* __restrict__ p, int ld,
                                          int row_base, int k0, int lane) {
    int m = lane & 15, hi = lane >> 4;
    const float* r = p + (size_t)(row_base + m) * ld + k0 + hi * 8;
    v4f a0 = *(const v4f*)(r);
    v4f a1 = *(const v4f*)(r + 4);
    v4f b0 = *(const v4f*)(r + 16);
    v4f b1 = *(const v4f*)(r + 20);
    v16bf f;
#pragma unroll
    for (int i = 0; i < 4; ++i) {
        f[i]      = (__bf16)a0[i];
        f[4 + i]  = (__bf16)a1[i];
        f[8 + i]  = (__bf16)b0[i];
        f[12 + i] = (__bf16)b1[i];
    }
    return f;
}

// ---------------- math helpers ----------------
__device__ __forceinline__ float sigmoidf_(float v) { return 1.0f / (1.0f + __expf(-v)); }
__device__ __forceinline__ float tanhf_(float v) {
    float t = __expf(-2.0f * fabsf(v));
    float r = (1.0f - t) / (1.0f + t);
    return copysignf(r, v);
}
__device__ __forceinline__ float eluf_(float v) { return v > 0.0f ? v : (__expf(v) - 1.0f); }
__device__ __forceinline__ float softplusf_(float v) {
    return v > 15.0f ? v : __logf(1.0f + __expf(v));
}

#if HAVE_TDM
// 1D contiguous TDM copy: nelem f32 from global 'gaddr' into LDS byte offset 'lds_off'.
// D# per ISA 8.3/8.4: group0 = {count=1 | lds_addr | global_addr | type=2},
// group1 = {data_size=2(4B); tensor_dim0=tile_dim0=nelem; tensor_dim1=tile_dim1=1; stride0=nelem}.
__device__ __forceinline__ void tdm_load_1d_f32(unsigned lds_off, const float* gaddr, unsigned nelem) {
    unsigned long long ga = (unsigned long long)(size_t)gaddr;
    u32x4 g0;
    g0[0] = 1u;                                   // count=1 (valid), no gather/iterate
    g0[1] = lds_off;                              // lds_addr (bytes)
    g0[2] = (unsigned)ga;                         // global_addr[95:64]
    g0[3] = (unsigned)(ga >> 32) | (2u << 30);    // global_addr[120:96] | type=2
    i32x8 g1;
    g1[0] = 0x20000;                              // data_size=2 -> 4 bytes
    g1[1] = (int)((nelem & 0xFFFFu) << 16);       // tensor_dim0[15:0] at bits 63:48
    g1[2] = (int)(((nelem >> 16) & 0xFFFFu) | (1u << 16)); // tensor_dim0[31:16] | tensor_dim1=1
    g1[3] = (int)((nelem & 0xFFFFu) << 16);       // tensor_dim1 hi=0 | tile_dim0=nelem
    g1[4] = 1;                                    // tile_dim1=1, tile_dim2=0
    g1[5] = (int)nelem;                           // tensor_dim0_stride low32
    g1[6] = 0;                                    // stride0 hi | stride1 lo
    g1[7] = 0;
    i32x4 gz = {0, 0, 0, 0};
#if __clang_major__ >= 23
    i32x8 gz8 = {0, 0, 0, 0, 0, 0, 0, 0};
    __builtin_amdgcn_tensor_load_to_lds(g0, g1, gz, gz, gz8, 0);
#else
    __builtin_amdgcn_tensor_load_to_lds(g0, g1, gz, gz, 0);
#endif
}
#endif

// ---------------- kernel 0: f32 -> bf16 convert of all 5 weight matrices ----------------
// segment boundaries (element counts): encw 8192 | wih 98304 | whh 196608 | muw 8192 | sigw 8192
__global__ void k_cvt_all(const float* __restrict__ s0, const float* __restrict__ s1,
                          const float* __restrict__ s2, const float* __restrict__ s3,
                          const float* __restrict__ s4, __bf16* __restrict__ dst) {
    int i = blockIdx.x * blockDim.x + threadIdx.x;   // 0 .. 319487
    const float* src;
    int off;
    if (i < 8192)                { src = s0; off = i; }
    else if (i < 8192 + 98304)   { src = s1; off = i - 8192; }
    else if (i < 106496 + 196608){ src = s2; off = i - 106496; }
    else if (i < 303104 + 8192)  { src = s3; off = i - 303104; }
    else                         { src = s4; off = i - 311296; }
    dst[i] = (__bf16)src[off];
}

// ---------------- kernel 1: encoder  enc = ELU(x @ enc_w^T + enc_b) ----------------
// each wave: 32 rows (two 16-row A tiles, B fragments reused 2x). 512 blocks x 8 waves.
__global__ __launch_bounds__(256) void k_encoder(const float* __restrict__ x,
                                                 const __bf16* __restrict__ encw,
                                                 const float* __restrict__ encb,
                                                 __bf16* __restrict__ enc_out) {
    int wave = blockIdx.x * 8 + (threadIdx.x >> 5);
    int lane = threadIdx.x & 31;
    int row0 = wave * 32;
    int hi = lane >> 4, n = lane & 15;

    v16bf a[2][2];
#pragma unroll
    for (int rt = 0; rt < 2; ++rt) {
        a[rt][0] = frag_f32(x, IND, row0 + 16 * rt, 0, lane);
        a[rt][1] = frag_f32(x, IND, row0 + 16 * rt, 32, lane);
    }

#pragma unroll 1
    for (int nt = 0; nt < HID / 16; ++nt) {
        int col0 = nt * 16;
        v16bf b0 = frag_bf16(encw, IND, col0, 0, lane);
        v16bf b1 = frag_bf16(encw, IND, col0, 32, lane);
        float bias = encb[col0 + n];
#pragma unroll
        for (int rt = 0; rt < 2; ++rt) {
            v8f c = {0,0,0,0,0,0,0,0};
            c = wmma_bf16(a[rt][0], b0, c);
            c = wmma_bf16(a[rt][1], b1, c);
#pragma unroll
            for (int r = 0; r < 8; ++r) {
                float v = eluf_(c[r] + bias);
                enc_out[(size_t)(row0 + 16 * rt + hi * 8 + r) * HID + col0 + n] = (__bf16)v;
            }
        }
    }
}

// ---------------- kernel 2: input gates  gx = enc @ w_ih^T + b_ih ----------------
// two 16-row A tiles per wave: w_ih fragments loaded once, used by 2 WMMAs.
__global__ __launch_bounds__(256) void k_gates(const __bf16* __restrict__ enc,
                                               const __bf16* __restrict__ wih,
                                               const float* __restrict__ bih,
                                               float* __restrict__ gx) {
    int wave = blockIdx.x * 8 + (threadIdx.x >> 5);
    int lane = threadIdx.x & 31;
    int row0 = wave * 32;
    int hi = lane >> 4, n = lane & 15;

    v16bf a[2][4];
#pragma unroll
    for (int rt = 0; rt < 2; ++rt)
#pragma unroll
        for (int k = 0; k < 4; ++k)
            a[rt][k] = frag_bf16(enc, HID, row0 + 16 * rt, 32 * k, lane);

#pragma unroll 1
    for (int nt = 0; nt < G3 / 16; ++nt) {
        int g0 = nt * 16;
        v8f c0 = {0,0,0,0,0,0,0,0};
        v8f c1 = {0,0,0,0,0,0,0,0};
#pragma unroll
        for (int k = 0; k < 4; ++k) {
            v16bf b = frag_bf16(wih, HID, g0, 32 * k, lane);
            c0 = wmma_bf16(a[0][k], b, c0);
            c1 = wmma_bf16(a[1][k], b, c1);
        }
        float bias = bih[g0 + n];
#pragma unroll
        for (int r = 0; r < 8; ++r) {
            gx[(size_t)(row0 + hi * 8 + r) * G3 + g0 + n]      = c0[r] + bias;
            gx[(size_t)(row0 + 16 + hi * 8 + r) * G3 + g0 + n] = c1[r] + bias;
        }
    }
}

// ---------------- kernel 3: persistent GRU recurrence ----------------
// 16 blocks (one per 16-batch-row tile), 8 waves, each wave owns 2 column tiles.
// Batch rows independent -> zero inter-block sync across all 512 steps.
// h f32 state in registers (C layout); bf16 copy double-buffered in LDS for A frags.
// gx slab (16 rows x 768 f32 = 48 KB) DMA'd into LDS per step by the Tensor Data
// Mover (wave 0 issues, TENSORcnt-tracked), overlapping the previous step's tail.
__global__ __launch_bounds__(256) void k_recur(const float* __restrict__ gx,
                                               const __bf16* __restrict__ whh,
                                               const float* __restrict__ bhh,
                                               float* __restrict__ hidden) {
    __shared__ __align__(16) __bf16 hbuf[2][16 * CELL];      // 16 KB
#if HAVE_TDM
    __shared__ __align__(16) float gxlds[16 * G3];           // 48 KB
#endif

    int lane = threadIdx.x & 31;
    int wv   = threadIdx.x >> 5;        // 0..7
    int row0 = blockIdx.x * 16;         // batch-row tile
    int hi = lane >> 4, n = lane & 15;

    v8f hprev[2] = {};                  // f32 h for this wave's two 16-col tiles

#if HAVE_TDM
    unsigned gx_lds_off = (unsigned)(unsigned long long)(size_t)&gxlds[0];
    if (wv == 0)
        tdm_load_1d_f32(gx_lds_off, gx + (size_t)row0 * G3, 16 * G3);   // slab for t=0
#endif

#pragma unroll 1
    for (int t = 0; t < T_; ++t) {
#if HAVE_TDM
        if (wv == 0) __builtin_amdgcn_s_wait_tensorcnt(0);   // gx[t] slab landed
        __syncthreads();                                     // visible to all waves
        const float* gxt = &gxlds[0];
#else
        // prefetch next step's gx slab for this wave's columns (global_prefetch_b8)
        if (t + 1 < T_) {
            const float* gx1 = gx + ((size_t)(t + 1) * B_ + row0) * G3 + wv * 32;
#pragma unroll
            for (int g = 0; g < 3; ++g)
                __builtin_prefetch(gx1 + (size_t)(lane & 15) * G3 + g * CELL + (lane >> 4) * 16, 0, 1);
        }
        const float* gxt = gx + ((size_t)t * B_ + row0) * G3;
#endif

        v16bf a[8] = {};
        if (t > 0) {
            const __bf16* hb = &hbuf[(t - 1) & 1][0];
#pragma unroll
            for (int k = 0; k < 8; ++k) a[k] = frag_bf16(hb, CELL, 0, 32 * k, lane);
        }

        __bf16* hout = &hbuf[t & 1][0];

#pragma unroll
        for (int ct = 0; ct < 2; ++ct) {
            int c0 = (wv * 2 + ct) * 16;        // h-column base, 0..240
            v8f gr = {0,0,0,0,0,0,0,0};
            v8f gz = {0,0,0,0,0,0,0,0};
            v8f gn = {0,0,0,0,0,0,0,0};
            if (t > 0) {
#pragma unroll
                for (int k = 0; k < 8; ++k) {
                    gr = wmma_bf16(a[k], frag_bf16(whh, CELL, c0,            32 * k, lane), gr);
                    gz = wmma_bf16(a[k], frag_bf16(whh, CELL, CELL + c0,     32 * k, lane), gz);
                    gn = wmma_bf16(a[k], frag_bf16(whh, CELL, 2 * CELL + c0, 32 * k, lane), gn);
                }
            }
            float br = bhh[c0 + n];
            float bz = bhh[CELL + c0 + n];
            float bn = bhh[2 * CELL + c0 + n];

            v8f hnew;
#pragma unroll
            for (int r = 0; r < 8; ++r) {
                int rr = hi * 8 + r;
                const float* grow = gxt + (size_t)rr * G3;
                float xr = grow[c0 + n];
                float xz = grow[CELL + c0 + n];
                float xn = grow[2 * CELL + c0 + n];
                float rg = sigmoidf_(xr + gr[r] + br);
                float zg = sigmoidf_(xz + gz[r] + bz);
                float ng = tanhf_(xn + rg * (gn[r] + bn));
                float h  = (1.0f - zg) * ng + zg * hprev[ct][r];
                hnew[r] = h;
                hidden[((size_t)t * B_ + row0 + rr) * CELL + c0 + n] = h;
            }
            hprev[ct] = hnew;
#pragma unroll
            for (int r = 0; r < 8; ++r)
                hout[(hi * 8 + r) * CELL + c0 + n] = (__bf16)hnew[r];
        }
        __syncthreads();                        // hbuf[t] complete; gxlds reads done
#if HAVE_TDM
        if (wv == 0 && t + 1 < T_)              // DMA next slab while step t+1 ramps
            tdm_load_1d_f32(gx_lds_off, gx + ((size_t)(t + 1) * B_ + row0) * G3, 16 * G3);
#endif
    }
}

// ---------------- kernel 4: output heads ----------------
// two 16-row A tiles per wave; mu pass then sig pass to bound register pressure.
__global__ __launch_bounds__(256) void k_heads(const float* __restrict__ hidden,
                                               const __bf16* __restrict__ muw,
                                               const float* __restrict__ mub,
                                               const __bf16* __restrict__ sigw,
                                               const float* __restrict__ sigb,
                                               float* __restrict__ mu,
                                               float* __restrict__ sig) {
    int wave = blockIdx.x * 8 + (threadIdx.x >> 5);
    int lane = threadIdx.x & 31;
    int row0 = wave * 32;
    int hi = lane >> 4, n = lane & 15;

    // ---- mu pass (hidden cols 0..127) ----
    {
        v16bf a[2][4];
#pragma unroll
        for (int rt = 0; rt < 2; ++rt)
#pragma unroll
            for (int k = 0; k < 4; ++k)
                a[rt][k] = frag_f32(hidden, CELL, row0 + 16 * rt, 32 * k, lane);
#pragma unroll 1
        for (int nt = 0; nt < OUTD / 16; ++nt) {
            int col0 = nt * 16;
            v8f c0 = {0,0,0,0,0,0,0,0};
            v8f c1 = {0,0,0,0,0,0,0,0};
#pragma unroll
            for (int k = 0; k < 4; ++k) {
                v16bf b = frag_bf16(muw, HID, col0, 32 * k, lane);
                c0 = wmma_bf16(a[0][k], b, c0);
                c1 = wmma_bf16(a[1][k], b, c1);
            }
            float bm = mub[col0 + n];
#pragma unroll
            for (int r = 0; r < 8; ++r) {
                mu[(size_t)(row0 + hi * 8 + r) * OUTD + col0 + n]      = eluf_(c0[r] + bm);
                mu[(size_t)(row0 + 16 + hi * 8 + r) * OUTD + col0 + n] = eluf_(c1[r] + bm);
            }
        }
    }
    // ---- sig pass (hidden cols 128..255) ----
    {
        v16bf a[2][4];
#pragma unroll
        for (int rt = 0; rt < 2; ++rt)
#pragma unroll
            for (int k = 0; k < 4; ++k)
                a[rt][k] = frag_f32(hidden, CELL, row0 + 16 * rt, HID + 32 * k, lane);
#pragma unroll 1
        for (int nt = 0; nt < OUTD / 16; ++nt) {
            int col0 = nt * 16;
            v8f c0 = {0,0,0,0,0,0,0,0};
            v8f c1 = {0,0,0,0,0,0,0,0};
#pragma unroll
            for (int k = 0; k < 4; ++k) {
                v16bf b = frag_bf16(sigw, HID, col0, 32 * k, lane);
                c0 = wmma_bf16(a[0][k], b, c0);
                c1 = wmma_bf16(a[1][k], b, c1);
            }
            float bs = sigb[col0 + n];
#pragma unroll
            for (int r = 0; r < 8; ++r) {
                sig[(size_t)(row0 + hi * 8 + r) * OUTD + col0 + n]      = softplusf_(c0[r] + bs) + 0.1f;
                sig[(size_t)(row0 + 16 + hi * 8 + r) * OUTD + col0 + n] = softplusf_(c1[r] + bs) + 0.1f;
            }
        }
    }
}

// ---------------- launch ----------------
extern "C" void kernel_launch(void* const* d_in, const int* in_sizes, int n_in,
                              void* d_out, int out_size, void* d_ws, size_t ws_size,
                              hipStream_t stream) {
    const float* x     = (const float*)d_in[0];
    const float* enc_w = (const float*)d_in[1];
    const float* enc_b = (const float*)d_in[2];
    const float* w_ih  = (const float*)d_in[3];
    const float* b_ih  = (const float*)d_in[4];
    const float* w_hh  = (const float*)d_in[5];
    const float* b_hh  = (const float*)d_in[6];
    const float* mu_w  = (const float*)d_in[7];
    const float* mu_b  = (const float*)d_in[8];
    const float* sig_w = (const float*)d_in[9];
    const float* sig_b = (const float*)d_in[10];

    // ws layout (all 16B aligned)
    char* ws = (char*)d_ws;
    float*  gx      = (float*)ws;                                   // 512*256*768*4 = 402653184 B
    __bf16* encbuf  = (__bf16*)(ws + 402653184ull);                 // 131072*128*2  = 33554432 B
    __bf16* encw_bf = (__bf16*)(ws + 402653184ull + 33554432ull);   // weights (bf16)
    __bf16* wih_bf  = encw_bf + HID * IND;        // +8192
    __bf16* whh_bf  = wih_bf  + G3 * HID;         // +98304
    __bf16* muw_bf  = whh_bf  + G3 * CELL;        // +196608
    __bf16* sigw_bf = muw_bf  + OUTD * HID;       // +8192

    float* mu     = (float*)d_out;
    float* sig    = mu  + (size_t)ROWS * OUTD;    // 8388608
    float* hidden = sig + (size_t)ROWS * OUTD;    // 8388608 -> hidden 33554432

    // 0) all weight conversions in one launch (319488 elements)
    k_cvt_all<<<(319488 + 255) / 256, 256, 0, stream>>>(enc_w, w_ih, w_hh, mu_w, sig_w, encw_bf);

    // 1) encoder (131072 rows / 32 rows-per-wave / 8 waves-per-block = 512 blocks)
    k_encoder<<<512, 256, 0, stream>>>(x, encw_bf, enc_b, encbuf);

    // 2) input-side gate pre-activations for all timesteps
    k_gates<<<512, 256, 0, stream>>>(encbuf, wih_bf, b_ih, gx);

    // 3) persistent recurrence: 16 independent batch-row tiles
    k_recur<<<16, 256, 0, stream>>>(gx, whh_bf, b_hh, hidden);

    // 4) output heads
    k_heads<<<512, 256, 0, stream>>>(hidden, muw_bf, mu_b, sigw_bf, sig_b, mu, sig);
}
